// SlotAttention_53480932769838
// MI455X (gfx1250) — matrix operand order repaired and checked by reading, not compiled, not run on hardware
//
#include <hip/hip_runtime.h>
#include <hip/hip_bf16.h>
#include <hip/hip_fp16.h>

typedef __attribute__((ext_vector_type(16))) _Float16 v16h;
typedef __attribute__((ext_vector_type(8)))  _Float16 v8h;
typedef __attribute__((ext_vector_type(8)))  float    v8f;

#define DEV static __device__ __forceinline__

constexpr int BATCH = 64;
constexpr int NTOK  = 784;   // tokens
constexpr int DIM   = 1024;  // feature dim
constexpr int NSLOT = 200;   // slots
constexpr int SP    = 224;   // padded slots (mult of 32)
constexpr int NP    = 832;   // padded tokens (mult of 64)
constexpr int GD3   = 3 * DIM;

// ---------------------------------------------------------------------------
// WMMA fragment helpers (f16 16x16x32, NT layout: both operands row-major with
// contiguous contraction dim). Per ISA 7.12.2:
//  A (16x32): lanes 0-15 row M=lane, VGPR0-3 = K[0..7], VGPR4-7 = K[16..23];
//             lanes 16-31 same rows, K[8..15] / K[24..31].
//  B (32x16): lanes 0-15 col N=lane hold K[0..7], K[16..23]; lanes 16-31 K[8..15], K[24..31].
// ---------------------------------------------------------------------------
DEV v16h load_a_frag(const _Float16* __restrict__ base, int lda, int lane) {
  int r    = lane & 15;
  int koff = (lane >> 4) << 3;                 // 0 or 8
  const _Float16* p = base + (size_t)r * lda + koff;
  v8h lo = *(const v8h*)(p);                   // K koff..koff+7
  v8h hi = *(const v8h*)(p + 16);              // K koff+16..koff+23
  v16h f;
#pragma unroll
  for (int i = 0; i < 8; ++i) { f[i] = lo[i]; f[i + 8] = hi[i]; }
  return f;
}

DEV v16h load_b_frag(const _Float16* __restrict__ base, int ldb, int lane) {
  int n    = lane & 15;
  int koff = (lane >> 4) << 3;                 // 0 or 8
  const _Float16* p = base + (size_t)n * ldb + koff;
  v8h lo = *(const v8h*)(p);
  v8h hi = *(const v8h*)(p + 16);
  v16h f;
#pragma unroll
  for (int i = 0; i < 8; ++i) { f[i] = lo[i]; f[i + 8] = hi[i]; }
  return f;
}

DEV v8f wmma_f16(v16h a, v16h b, v8f c) {
  return __builtin_amdgcn_wmma_f32_16x16x32_f16(false, a, false, b,
                                                (short)0, c, false, false);
}

// Generic per-wave 32x64 NT GEMM: C = A(MxK) * B(NxK)^T, f32 accumulate.
// Loads all fragments of a K-step before issuing WMMAs so the scheduler can
// cover many loads with one partial s_wait_loadcnt.
template <typename Epi>
DEV void gemm_wave_32x64(const _Float16* __restrict__ A, int lda,
                         const _Float16* __restrict__ B, int ldb,
                         int K, int row0, int col0, int lane, Epi epi) {
  v8f acc[2][4] = {};
  const _Float16* a0 = A + (size_t)row0 * lda;
  const _Float16* a1 = a0 + (size_t)16 * lda;
  const _Float16* b0 = B + (size_t)col0 * ldb;
  for (int k = 0; k < K; k += 32) {
    v16h af0 = load_a_frag(a0 + k, lda, lane);
    v16h af1 = load_a_frag(a1 + k, lda, lane);
    v16h bf[4];
#pragma unroll
    for (int j = 0; j < 4; ++j)
      bf[j] = load_b_frag(b0 + (size_t)(j * 16) * ldb + k, ldb, lane);
#pragma unroll
    for (int j = 0; j < 4; ++j) {
      acc[0][j] = wmma_f16(af0, bf[j], acc[0][j]);
      acc[1][j] = wmma_f16(af1, bf[j], acc[1][j]);
    }
  }
  // C layout: VGPR r, lanes 0-15 -> (row0+r, col0+lane); lanes 16-31 -> (+8, -16)
  int col   = lane & 15;
  int rbase = (lane >> 4) << 3;
#pragma unroll
  for (int i = 0; i < 2; ++i)
#pragma unroll
    for (int j = 0; j < 4; ++j)
#pragma unroll
      for (int r = 0; r < 8; ++r)
        epi(row0 + i * 16 + rbase + r, col0 + j * 16 + col, acc[i][j][r]);
}

// ---------------------------------------------------------------------------
// Utility kernels
// ---------------------------------------------------------------------------
__global__ void f32_to_f16_kernel(const float* __restrict__ in,
                                  _Float16* __restrict__ out, int n) {
  int i = blockIdx.x * 256 + threadIdx.x;
  if (i < n) out[i] = (_Float16)in[i];
}

__global__ void zero_loss_kernel(float* p) {
  if (threadIdx.x == 0 && blockIdx.x == 0) *p = 0.f;
}

__global__ void zero_kpad_kernel(_Float16* __restrict__ K) {
  // zero k rows NTOK..NP-1 per batch
  int i = blockIdx.x * 256 + threadIdx.x;          // BATCH*(NP-NTOK)*DIM
  const int tot = BATCH * (NP - NTOK) * DIM;
  if (i >= tot) return;
  int d = i & (DIM - 1);
  int rest = i >> 10;
  int jp = rest % (NP - NTOK);
  int b  = rest / (NP - NTOK);
  K[((size_t)b * NP + NTOK + jp) * DIM + d] = (_Float16)0.f;
}

__global__ void slots_init_kernel(const float* __restrict__ init_slots,
                                  float* __restrict__ Sf,
                                  _Float16* __restrict__ Sh) {
  int i = blockIdx.x * 256 + threadIdx.x;          // BATCH*SP*DIM
  const int tot = BATCH * SP * DIM;
  if (i >= tot) return;
  int d = i & (DIM - 1);
  int s = (i >> 10) % SP;
  float v = (s < NSLOT) ? init_slots[(size_t)s * DIM + d] : 0.f;
  Sf[i] = v;
  Sh[i] = (_Float16)v;
}

// Transpose inputs_x (B,N,D) f32 -> xT (B,D,NP) f16, zero-padding tokens.
__global__ void transpose_x_kernel(const float* __restrict__ X,
                                   _Float16* __restrict__ XT) {
  __shared__ _Float16 tile[32][33];
  int b  = blockIdx.z;
  int j0 = blockIdx.x * 32;                        // token tile (26 tiles -> NP)
  int d0 = blockIdx.y * 32;                        // dim tile
  const float* Xb = X + (size_t)b * NTOK * DIM;
  _Float16* Tb = XT + (size_t)b * DIM * NP;
  int tx = threadIdx.x, ty = threadIdx.y;          // 32 x 8
#pragma unroll
  for (int r = 0; r < 32; r += 8) {
    int j = j0 + ty + r;
    float v = (j < NTOK) ? Xb[(size_t)j * DIM + d0 + tx] : 0.f;
    tile[ty + r][tx] = (_Float16)v;
  }
  __syncthreads();
#pragma unroll
  for (int r = 0; r < 32; r += 8) {
    int d = d0 + ty + r;
    Tb[(size_t)d * NP + j0 + tx] = tile[tx][ty + r];
  }
}

// ---------------------------------------------------------------------------
// GEMM kernels
// ---------------------------------------------------------------------------
// k = inputs @ k_w^T + k_b ; output f16 into padded (B, NP, DIM) layout.
// Padded row index: b*NP + (m - b*NTOK) = m + (NP-NTOK)*b = m + 48*b, and a
// 32-row tile crosses at most one batch boundary -> one compare per element.
__global__ void kproj_kernel(const _Float16* __restrict__ Ain,
                             const _Float16* __restrict__ Wk,
                             const float* __restrict__ kb,
                             _Float16* __restrict__ Kout) {
  int lane = threadIdx.x;
  int wid  = blockIdx.x * blockDim.y + threadIdx.y;
  const int tilesN = DIM / 64;                     // 16
  const int tilesM = (BATCH * NTOK) / 32;          // 1568
  if (wid >= tilesM * tilesN) return;
  int tm = wid / tilesN, tn = wid % tilesN;
  int row0  = tm * 32;
  int b0    = row0 / NTOK;
  int bound = (b0 + 1) * NTOK;
  gemm_wave_32x64(Ain, DIM, Wk, DIM, DIM, row0, tn * 64, lane,
      [&](int m, int n, float v) {
        int b = (m >= bound) ? (b0 + 1) : b0;
        Kout[(size_t)(m + (NP - NTOK) * b) * DIM + n] = (_Float16)(v + kb[n]);
      });
}

// dots = (slots @ k^T) * scale ; per-batch (SP x NP) f32
__global__ void dots_kernel(const _Float16* __restrict__ Sh,
                            const _Float16* __restrict__ Kf,
                            float* __restrict__ dots, float scale) {
  int lane = threadIdx.x;
  int wid  = blockIdx.x * blockDim.y + threadIdx.y;
  const int tiles = (SP / 32) * (NP / 64);         // 7*13 = 91
  if (wid >= BATCH * tiles) return;
  int b = wid / tiles, t = wid % tiles;
  int tm = t / (NP / 64), tn = t % (NP / 64);
  const _Float16* A = Sh + (size_t)b * SP * DIM;
  const _Float16* B = Kf + (size_t)b * NP * DIM;
  float* C = dots + (size_t)b * SP * NP;
  gemm_wave_32x64(A, DIM, B, DIM, DIM, tm * 32, tn * 64, lane,
      [&](int m, int n, float v) { C[(size_t)m * NP + n] = v * scale; });
}

// updates = (attn @ x) / NTOK ; per-batch (SP x DIM), writes f32 + f16
__global__ void updates_kernel(const _Float16* __restrict__ Attn,
                               const _Float16* __restrict__ XT,
                               float* __restrict__ Uf,
                               _Float16* __restrict__ Uh, float invN) {
  int lane = threadIdx.x;
  int wid  = blockIdx.x * blockDim.y + threadIdx.y;
  const int tiles = (SP / 32) * (DIM / 64);        // 7*16 = 112
  if (wid >= BATCH * tiles) return;
  int b = wid / tiles, t = wid % tiles;
  int tm = t / (DIM / 64), tn = t % (DIM / 64);
  const _Float16* A = Attn + (size_t)b * SP * NP;
  const _Float16* B = XT + (size_t)b * DIM * NP;
  size_t obase = (size_t)b * SP * DIM;
  gemm_wave_32x64(A, NP, B, NP, NP, tm * 32, tn * 64, lane,
      [&](int m, int n, float v) {
        float u = v * invN;
        size_t o = obase + (size_t)m * DIM + n;
        Uf[o] = u;
        Uh[o] = (_Float16)u;
      });
}

// Fused GRU: per wave one 16(slots) x 16(dim) tile; 6 gate accumulators.
__global__ void gru_kernel(const _Float16* __restrict__ U,
                           const _Float16* __restrict__ Shc,
                           const _Float16* __restrict__ Wih,
                           const _Float16* __restrict__ Whh,
                           const float* __restrict__ bih,
                           const float* __restrict__ bhh,
                           float* __restrict__ Sf,
                           _Float16* __restrict__ Shn) {
  int lane = threadIdx.x;
  int wid  = blockIdx.x * blockDim.y + threadIdx.y;
  const int tiles = (SP / 16) * (DIM / 16);        // 14*64 = 896
  if (wid >= BATCH * tiles) return;
  int b = wid / tiles, t = wid % tiles;
  int tm = t / (DIM / 16), td = t % (DIM / 16);
  int row0 = tm * 16, d0 = td * 16;
  const _Float16* Ab = U   + (size_t)b * SP * DIM + (size_t)row0 * DIM;
  const _Float16* Hb = Shc + (size_t)b * SP * DIM + (size_t)row0 * DIM;
  v8f acc[6] = {};                                 // ir iz in hr hz hn
  for (int k = 0; k < DIM; k += 32) {
    v16h au = load_a_frag(Ab + k, DIM, lane);
    v16h ah = load_a_frag(Hb + k, DIM, lane);
    v16h bi[3], bh[3];
#pragma unroll
    for (int g = 0; g < 3; ++g) {
      bi[g] = load_b_frag(Wih + ((size_t)(g * DIM + d0)) * DIM + k, DIM, lane);
      bh[g] = load_b_frag(Whh + ((size_t)(g * DIM + d0)) * DIM + k, DIM, lane);
    }
#pragma unroll
    for (int g = 0; g < 3; ++g) {
      acc[g]     = wmma_f16(au, bi[g], acc[g]);
      acc[3 + g] = wmma_f16(ah, bh[g], acc[3 + g]);
    }
  }
  int col   = lane & 15;
  int rbase = (lane >> 4) << 3;
  int d = d0 + col;
  float bir = bih[d], biz = bih[DIM + d], bin = bih[2 * DIM + d];
  float bhr = bhh[d], bhz = bhh[DIM + d], bhn = bhh[2 * DIM + d];
#pragma unroll
  for (int r = 0; r < 8; ++r) {
    int s = row0 + rbase + r;
    size_t o = (size_t)b * SP * DIM + (size_t)s * DIM + d;
    float ir = acc[0][r] + bir, iz = acc[1][r] + biz, in_ = acc[2][r] + bin;
    float hr = acc[3][r] + bhr, hz = acc[4][r] + bhz, hn = acc[5][r] + bhn;
    float rv = 1.f / (1.f + __expf(-(ir + hr)));
    float zv = 1.f / (1.f + __expf(-(iz + hz)));
    float nv = tanhf(in_ + rv * hn);
    float hold = Sf[o];
    float hnew = (1.f - zv) * nv + zv * hold;
    Sf[o] = hnew;
    Shn[o] = (_Float16)hnew;
  }
}

// ---------------------------------------------------------------------------
// Normalization / reductions
// ---------------------------------------------------------------------------
__global__ void rowsum_kernel(const float* __restrict__ dots,
                              float* __restrict__ rowsum) {
  int b = blockIdx.x / NSLOT, s = blockIdx.x % NSLOT;
  const float* row = dots + ((size_t)b * SP + s) * NP;
  float acc = 0.f;
  for (int n = threadIdx.x; n < NP; n += 256) acc += row[n];
  __shared__ float red[256];
  red[threadIdx.x] = acc;
  __syncthreads();
  for (int st = 128; st > 0; st >>= 1) {
    if (threadIdx.x < st) red[threadIdx.x] += red[threadIdx.x + st];
    __syncthreads();
  }
  if (threadIdx.x == 0) rowsum[blockIdx.x] = red[0];
}

__global__ void total_kernel(const float* __restrict__ rowsum,
                             float* __restrict__ total) {
  int b = blockIdx.x;
  float acc = (threadIdx.x < NSLOT) ? rowsum[b * NSLOT + threadIdx.x] : 0.f;
  __shared__ float red[256];
  red[threadIdx.x] = acc;
  __syncthreads();
  for (int st = 128; st > 0; st >>= 1) {
    if (threadIdx.x < st) red[threadIdx.x] += red[threadIdx.x + st];
    __syncthreads();
  }
  if (threadIdx.x == 0) total[b] = red[0];
}

__global__ void attn_kernel(const float* __restrict__ dots,
                            const float* __restrict__ rowsum,
                            const float* __restrict__ total,
                            _Float16* __restrict__ attn,
                            float* __restrict__ loss, int accum_loss) {
  size_t i = (size_t)blockIdx.x * 256 + threadIdx.x;
  const size_t TOT = (size_t)BATCH * SP * NP;
  float a = 0.f;
  if (i < TOT) {
    int n = (int)(i % NP);
    int s = (int)((i / NP) % SP);
    int b = (int)(i / ((size_t)NP * SP));
    if (s < NSLOT && n < NTOK) {
      float v = dots[i] * total[b] / rowsum[b * NSLOT + s];
      a = 1.f / (1.f + __expf(-v));
    }
    attn[i] = (_Float16)a;
  }
  if (accum_loss) {
    __shared__ float red[256];
    red[threadIdx.x] = a;
    __syncthreads();
    for (int st = 128; st > 0; st >>= 1) {
      if (threadIdx.x < st) red[threadIdx.x] += red[threadIdx.x + st];
      __syncthreads();
    }
    if (threadIdx.x == 0) atomicAdd(loss, red[0]);
  }
}

__global__ void logits_kernel(const float* __restrict__ U,
                              const float* __restrict__ loss,
                              float* __restrict__ out) {
  int b = blockIdx.x / NSLOT, s = blockIdx.x % NSLOT;
  const float* row = U + ((size_t)b * SP + s) * DIM;
  float acc = 0.f;
  for (int d = threadIdx.x; d < DIM; d += 256) acc += row[d];
  __shared__ float red[256];
  red[threadIdx.x] = acc;
  __syncthreads();
  for (int st = 128; st > 0; st >>= 1) {
    if (threadIdx.x < st) red[threadIdx.x] += red[threadIdx.x + st];
    __syncthreads();
  }
  if (threadIdx.x == 0) out[blockIdx.x] = red[0];
  if (blockIdx.x == 0 && threadIdx.x == 0)
    out[BATCH * NSLOT] = (*loss) / ((float)BATCH * NSLOT * NTOK);
}

// ---------------------------------------------------------------------------
// Host launch
// ---------------------------------------------------------------------------
extern "C" void kernel_launch(void* const* d_in, const int* in_sizes, int n_in,
                              void* d_out, int out_size, void* d_ws, size_t ws_size,
                              hipStream_t stream) {
  const float* inputs        = (const float*)d_in[0];
  const float* inputs_x      = (const float*)d_in[1];
  const float* initial_slots = (const float*)d_in[2];
  const float* k_w           = (const float*)d_in[3];
  const float* k_b           = (const float*)d_in[4];
  const float* w_ih          = (const float*)d_in[5];
  const float* w_hh          = (const float*)d_in[6];
  const float* b_ih          = (const float*)d_in[7];
  const float* b_hh          = (const float*)d_in[8];
  float* out = (float*)d_out;

  char* ws = (char*)d_ws;
  size_t off = 0;
  auto alloc = [&](size_t bytes) -> char* {
    char* p = ws + off;
    off = (off + bytes + 255) & ~(size_t)255;
    return p;
  };
  _Float16* inF16  = (_Float16*)alloc((size_t)BATCH * NTOK * DIM * 2);
  _Float16* xT     = (_Float16*)alloc((size_t)BATCH * DIM * NP * 2);
  _Float16* kF16   = (_Float16*)alloc((size_t)BATCH * NP * DIM * 2);
  _Float16* kwF16  = (_Float16*)alloc((size_t)DIM * DIM * 2);
  _Float16* wihF16 = (_Float16*)alloc((size_t)GD3 * DIM * 2);
  _Float16* whhF16 = (_Float16*)alloc((size_t)GD3 * DIM * 2);
  float*    slotsF = (float*)   alloc((size_t)BATCH * SP * DIM * 4);
  _Float16* shA    = (_Float16*)alloc((size_t)BATCH * SP * DIM * 2);
  _Float16* shB    = (_Float16*)alloc((size_t)BATCH * SP * DIM * 2);
  float*    updF   = (float*)   alloc((size_t)BATCH * SP * DIM * 4);
  _Float16* updH   = (_Float16*)alloc((size_t)BATCH * SP * DIM * 2);
  float*    dotsF  = (float*)   alloc((size_t)BATCH * SP * NP * 4);
  _Float16* attnH  = (_Float16*)alloc((size_t)BATCH * SP * NP * 2);
  float*    rowsum = (float*)   alloc((size_t)BATCH * NSLOT * 4);
  float*    total  = (float*)   alloc((size_t)BATCH * 4);
  float*    lossA  = (float*)   alloc(4);

  const dim3 gblk(32, 8);  // 8 waves / block

  zero_loss_kernel<<<1, 32, 0, stream>>>(lossA);

  // Precision conversions
  f32_to_f16_kernel<<<(DIM * DIM + 255) / 256, 256, 0, stream>>>(k_w, kwF16, DIM * DIM);
  f32_to_f16_kernel<<<(GD3 * DIM + 255) / 256, 256, 0, stream>>>(w_ih, wihF16, GD3 * DIM);
  f32_to_f16_kernel<<<(GD3 * DIM + 255) / 256, 256, 0, stream>>>(w_hh, whhF16, GD3 * DIM);
  {
    int n = BATCH * NTOK * DIM;
    f32_to_f16_kernel<<<(n + 255) / 256, 256, 0, stream>>>(inputs, inF16, n);
  }
  transpose_x_kernel<<<dim3(NP / 32, DIM / 32, BATCH), dim3(32, 8), 0, stream>>>(inputs_x, xT);
  slots_init_kernel<<<(BATCH * SP * DIM + 255) / 256, 256, 0, stream>>>(initial_slots, slotsF, shA);
  zero_kpad_kernel<<<(BATCH * (NP - NTOK) * DIM + 255) / 256, 256, 0, stream>>>(kF16);

  // k projection: waves = 1568 * 16 = 25088 -> 3136 blocks
  kproj_kernel<<<(1568 * 16) / 8, gblk, 0, stream>>>(inF16, kwF16, k_b, kF16);

  const float scale = 0.03125f;             // 1024^-0.5
  const float invN  = 1.f / (float)NTOK;

  for (int iter = 0; iter < 3; ++iter) {
    _Float16* shCur = (iter & 1) ? shB : shA;
    _Float16* shNxt = (iter & 1) ? shA : shB;

    dots_kernel<<<(BATCH * 91) / 8, gblk, 0, stream>>>(shCur, kF16, dotsF, scale);
    rowsum_kernel<<<BATCH * NSLOT, 256, 0, stream>>>(dotsF, rowsum);
    total_kernel<<<BATCH, 256, 0, stream>>>(rowsum, total);
    attn_kernel<<<(BATCH * SP * NP) / 256, 256, 0, stream>>>(
        dotsF, rowsum, total, attnH, lossA, iter == 2 ? 1 : 0);
    updates_kernel<<<(BATCH * 112) / 8, gblk, 0, stream>>>(attnH, xT, updF, updH, invN);
    gru_kernel<<<(BATCH * 896) / 8, gblk, 0, stream>>>(
        updH, shCur, wihF16, whhF16, b_ih, b_hh, slotsF, shNxt);
  }

  logits_kernel<<<BATCH * NSLOT, 256, 0, stream>>>(updF, lossA, out);
}
